// SAGE_5927054868969
// MI455X (gfx1250) — compile-verified, matrix operand order checked by reading, not run on hardware
//
#include <hip/hip_runtime.h>
#include <hip/hip_bf16.h>

typedef __attribute__((ext_vector_type(16))) __bf16        v16bf;
typedef __attribute__((ext_vector_type(8)))  float         v8f;
typedef __attribute__((ext_vector_type(8)))  unsigned int  v8u;

#define N_NODES 50000
#define N_EDGES 800000
#define CH      128
#define NGRAPH  512
#define BN_EPS  1e-5f

// ---------- f32 -> bf16 (round-to-nearest-even), used only in one-pass cvt kernels ----------
__device__ __forceinline__ unsigned f2bf1(float f) {
    unsigned u = __float_as_uint(f);
    u += 0x7FFFu + ((u >> 16) & 1u);
    return u >> 16;
}
__device__ __forceinline__ unsigned pack2(float lo, float hi) {
    return f2bf1(lo) | (f2bf1(hi) << 16);
}

// Load a 16x32 bf16 WMMA operand fragment from a pre-converted bf16 row.
// Lane owns row m=lane&15 (A) or W row n=lane&15 (feeding B = W^T columns).
// kh = lane>>4: VGPR0..3 hold K-octet kh*8, VGPR4..7 hold K-octet 16+kh*8.
// Two 16-byte loads -> global_load_b128 each.
__device__ __forceinline__ v16bf load_frag_bf(const unsigned short* __restrict__ rowPtr,
                                              int kbase, int kh) {
    const uint4 a = *(const uint4*)(rowPtr + kbase + kh * 8);
    const uint4 b = *(const uint4*)(rowPtr + kbase + 16 + kh * 8);
    v8u pk;
    pk[0] = a.x; pk[1] = a.y; pk[2] = a.z; pk[3] = a.w;
    pk[4] = b.x; pk[5] = b.y; pk[6] = b.z; pk[7] = b.w;
    return __builtin_bit_cast(v16bf, pk);
}

// ---------------- zero fill ----------------
__global__ void zero_kernel(float* __restrict__ p, size_t n) {
    size_t i = (size_t)blockIdx.x * blockDim.x + threadIdx.x;
    size_t stride = (size_t)gridDim.x * blockDim.x;
    for (; i < n; i += stride) p[i] = 0.0f;
}

// ---------------- f32[rows,128] -> bf16[rows,128], optional per-row 1/max(deg,1) ----------------
// one thread per bf16 pair; row = pairIndex >> 6 since 128/2 = 64 pairs per row.
__global__ void cvt_bf16_kernel(const float* __restrict__ in, const float* __restrict__ deg,
                                unsigned* __restrict__ out, int rows) {
    size_t n = (size_t)rows * (CH / 2);
    size_t stride = (size_t)gridDim.x * blockDim.x;
    for (size_t i = (size_t)blockIdx.x * blockDim.x + threadIdx.x; i < n; i += stride) {
        float s = 1.0f;
        if (deg != nullptr) {
            int row = (int)(i >> 6);
            s = 1.0f / fmaxf(deg[row], 1.0f);
        }
        float2 v = ((const float2*)in)[i];
        out[i] = pack2(v.x * s, v.y * s);
    }
}

// ---------------- edge scatter: agg[dst] += feat[src]; deg[dst] += 1 ----------------
// one wave per edge, 4 channels/lane: global_load_b128 + 4x global_atomic_add_f32 (L2-resident).
__global__ __launch_bounds__(256) void scatter_kernel(
    const long long* __restrict__ src, const long long* __restrict__ dst,
    const float* __restrict__ feat, float* __restrict__ agg,
    float* __restrict__ deg, int nEdges)
{
    int tid  = blockIdx.x * blockDim.x + threadIdx.x;
    int e    = tid >> 5;
    int lane = tid & 31;
    if (e >= nEdges) return;
    int s = (int)src[e];
    int d = (int)dst[e];
    const float4 v = *(const float4*)(feat + (size_t)s * CH + lane * 4);
    float* ap = agg + (size_t)d * CH + lane * 4;
    atomicAdd(ap + 0, v.x);
    atomicAdd(ap + 1, v.y);
    atomicAdd(ap + 2, v.z);
    atomicAdd(ap + 3, v.w);
    if (deg != nullptr && lane == 0) atomicAdd(&deg[d], 1.0f);
}

// ---------------- SAGE layer GEMM (pre-converted bf16 operands) ----------------
// out[M,128] = meanb @ Wl^T + featb @ Wr^T + bias
// one wave per 16x64 output strip: A fragment loaded once, reused for 4 WMMAs.
__global__ __launch_bounds__(128) void sage_gemm_kernel(
    const unsigned short* __restrict__ Ab1, const unsigned short* __restrict__ Wlb,
    const unsigned short* __restrict__ Ab2, const unsigned short* __restrict__ Wrb,
    const float* __restrict__ bias, float* __restrict__ out, int M)
{
    const int wave = (blockIdx.x * blockDim.x + threadIdx.x) >> 5;
    const int lane = threadIdx.x & 31;
    const int nWaves = (M >> 4) * 2;           // 2 column groups of 64
    if (wave >= nWaves) return;
    const int rowTile = wave >> 1;
    const int cg = wave & 1;
    const int r  = lane & 15;
    const int kh = lane >> 4;

    const unsigned short* aRow1 = Ab1 + (size_t)(rowTile * 16 + r) * CH;
    const unsigned short* aRow2 = Ab2 + (size_t)(rowTile * 16 + r) * CH;
    const unsigned short* wl    = Wlb + (size_t)(cg * 64 + r) * CH;
    const unsigned short* wr    = Wrb + (size_t)(cg * 64 + r) * CH;

    v8f acc[4] = {};
#pragma unroll
    for (int kb = 0; kb < CH; kb += 32) {
        v16bf a = load_frag_bf(aRow1, kb, kh);
#pragma unroll
        for (int sub = 0; sub < 4; sub++) {
            v16bf b = load_frag_bf(wl + (size_t)sub * 16 * CH, kb, kh);
            acc[sub] = __builtin_amdgcn_wmma_f32_16x16x32_bf16(false, a, false, b,
                                                               (short)0, acc[sub], false, false);
        }
    }
#pragma unroll
    for (int kb = 0; kb < CH; kb += 32) {
        v16bf a = load_frag_bf(aRow2, kb, kh);
#pragma unroll
        for (int sub = 0; sub < 4; sub++) {
            v16bf b = load_frag_bf(wr + (size_t)sub * 16 * CH, kb, kh);
            acc[sub] = __builtin_amdgcn_wmma_f32_16x16x32_bf16(false, a, false, b,
                                                               (short)0, acc[sub], false, false);
        }
    }

#pragma unroll
    for (int sub = 0; sub < 4; sub++) {
        const int col = cg * 64 + sub * 16 + r;      // D: lane holds N = lane&15
        const float bc = bias[col];
        float* op = out + (size_t)(rowTile * 16 + kh * 8) * CH + col;
#pragma unroll
        for (int i = 0; i < 8; i++) op[(size_t)i * CH] = acc[sub][i] + bc;
    }
}

// ---------------- global add pool: pooled[batch[n]] += h[n] ----------------
__global__ __launch_bounds__(256) void pool_kernel(
    const float* __restrict__ h, const long long* __restrict__ batch,
    float* __restrict__ pooled, int nNodes)
{
    int tid  = blockIdx.x * blockDim.x + threadIdx.x;
    int node = tid >> 5;
    int lane = tid & 31;
    if (node >= nNodes) return;
    int g = (int)batch[node];
    const float4 v = *(const float4*)(h + (size_t)node * CH + lane * 4);
    float* pp = pooled + (size_t)g * CH + lane * 4;
    atomicAdd(pp + 0, v.x);
    atomicAdd(pp + 1, v.y);
    atomicAdd(pp + 2, v.z);
    atomicAdd(pp + 3, v.w);
}

// ---------------- lin1 + BatchNorm(eval) + ReLU (WMMA, bf16 operands) ----------------
__global__ __launch_bounds__(128) void lin1_bn_relu_kernel(
    const unsigned short* __restrict__ Ab, const unsigned short* __restrict__ Wb,
    const float* __restrict__ bias,
    const float* __restrict__ gamma, const float* __restrict__ beta,
    const float* __restrict__ rmean, const float* __restrict__ rvar,
    float* __restrict__ out, int M)
{
    const int wave = (blockIdx.x * blockDim.x + threadIdx.x) >> 5;
    const int lane = threadIdx.x & 31;
    const int nWaves = (M >> 4) * 2;
    if (wave >= nWaves) return;
    const int rowTile = wave >> 1;
    const int cg = wave & 1;
    const int r  = lane & 15;
    const int kh = lane >> 4;

    const unsigned short* aRow = Ab + (size_t)(rowTile * 16 + r) * CH;
    const unsigned short* w    = Wb + (size_t)(cg * 64 + r) * CH;

    v8f acc[4] = {};
#pragma unroll
    for (int kb = 0; kb < CH; kb += 32) {
        v16bf a = load_frag_bf(aRow, kb, kh);
#pragma unroll
        for (int sub = 0; sub < 4; sub++) {
            v16bf b = load_frag_bf(w + (size_t)sub * 16 * CH, kb, kh);
            acc[sub] = __builtin_amdgcn_wmma_f32_16x16x32_bf16(false, a, false, b,
                                                               (short)0, acc[sub], false, false);
        }
    }

#pragma unroll
    for (int sub = 0; sub < 4; sub++) {
        const int col = cg * 64 + sub * 16 + r;
        const float s  = gamma[col] * rsqrtf(rvar[col] + BN_EPS);
        const float sh = beta[col] - rmean[col] * s;
        const float bc = bias[col];
        float* op = out + (size_t)(rowTile * 16 + kh * 8) * CH + col;
#pragma unroll
        for (int i = 0; i < 8; i++)
            op[(size_t)i * CH] = fmaxf((acc[sub][i] + bc) * s + sh, 0.0f);
    }
}

// ---------------- lin2: [512,128] @ [2,128]^T -> [512,2] ----------------
__global__ void lin2_kernel(const float* __restrict__ h, const float* __restrict__ w,
                            const float* __restrict__ b, float* __restrict__ out)
{
    int t = blockIdx.x * blockDim.x + threadIdx.x;
    if (t >= NGRAPH * 2) return;
    int g = t >> 1, o = t & 1;
    const float* hr = h + (size_t)g * CH;
    const float* wr = w + (size_t)o * CH;
    float s = 0.0f;
#pragma unroll 8
    for (int c = 0; c < CH; c++) s = fmaf(hr[c], wr[c], s);
    out[t] = s + b[o];
}

extern "C" void kernel_launch(void* const* d_in, const int* in_sizes, int n_in,
                              void* d_out, int out_size, void* d_ws, size_t ws_size,
                              hipStream_t stream) {
    const float*     x     = (const float*)d_in[0];
    const long long* ei    = (const long long*)d_in[1];   // int64 [2, E]
    const long long* batch = (const long long*)d_in[2];   // int64 [N]
    const float* W1l  = (const float*)d_in[3];
    const float* b1   = (const float*)d_in[4];
    const float* W1r  = (const float*)d_in[5];
    const float* W2l  = (const float*)d_in[6];
    const float* b2   = (const float*)d_in[7];
    const float* W2r  = (const float*)d_in[8];
    const float* l1w  = (const float*)d_in[9];
    const float* l1b  = (const float*)d_in[10];
    const float* l2w  = (const float*)d_in[11];
    const float* l2b  = (const float*)d_in[12];
    const float* gam  = (const float*)d_in[13];
    const float* bet  = (const float*)d_in[14];
    const float* rmean= (const float*)d_in[15];
    const float* rvar = (const float*)d_in[16];
    (void)in_sizes; (void)n_in; (void)out_size; (void)ws_size;

    const long long* src = ei;             // edge_index[0]
    const long long* dst = ei + N_EDGES;   // edge_index[1]

    // ---- workspace layout ----
    char* ws = (char*)d_ws;
    size_t off = 0;
    auto alloc = [&](size_t bytes) { void* p = ws + off; off += (bytes + 255) & ~(size_t)255; return p; };
    float* deg    = (float*)alloc((size_t)N_NODES * 4);
    float* agg    = (float*)alloc((size_t)N_NODES * CH * 4);
    float* h1     = (float*)alloc((size_t)N_NODES * CH * 4);
    float* h2     = (float*)alloc((size_t)N_NODES * CH * 4);
    float* pooled = (float*)alloc((size_t)NGRAPH * CH * 4);
    float* hg     = (float*)alloc((size_t)NGRAPH * CH * 4);
    unsigned short* aggb    = (unsigned short*)alloc((size_t)N_NODES * CH * 2);
    unsigned short* featb   = (unsigned short*)alloc((size_t)N_NODES * CH * 2);
    unsigned short* wlb     = (unsigned short*)alloc((size_t)CH * CH * 2);
    unsigned short* wrb     = (unsigned short*)alloc((size_t)CH * CH * 2);
    unsigned short* l1wb    = (unsigned short*)alloc((size_t)CH * CH * 2);
    unsigned short* pooledb = (unsigned short*)alloc((size_t)NGRAPH * CH * 2);

    const int zeroBlocks    = 1024;
    const int cvtBlocks     = 1024;
    const int scatterBlocks = (N_EDGES * 32 + 255) / 256;
    const int gemmBlocks    = ((N_NODES / 16) * 2 + 3) / 4;   // 4 waves/block
    const int poolBlocks    = (N_NODES * 32 + 255) / 256;
    const int headBlocks    = ((NGRAPH / 16) * 2 + 3) / 4;
    const int wcvtBlocks    = (CH * CH / 2 + 255) / 256;

    // ---- SAGEConv 1 ----
    zero_kernel<<<zeroBlocks, 256, 0, stream>>>(deg, (size_t)N_NODES);
    zero_kernel<<<zeroBlocks, 256, 0, stream>>>(agg, (size_t)N_NODES * CH);
    scatter_kernel<<<scatterBlocks, 256, 0, stream>>>(src, dst, x, agg, deg, N_EDGES);
    cvt_bf16_kernel<<<cvtBlocks, 256, 0, stream>>>(x,   nullptr, (unsigned*)featb, N_NODES);
    cvt_bf16_kernel<<<cvtBlocks, 256, 0, stream>>>(agg, deg,     (unsigned*)aggb,  N_NODES);
    cvt_bf16_kernel<<<wcvtBlocks, 256, 0, stream>>>(W1l, nullptr, (unsigned*)wlb, CH);
    cvt_bf16_kernel<<<wcvtBlocks, 256, 0, stream>>>(W1r, nullptr, (unsigned*)wrb, CH);
    sage_gemm_kernel<<<gemmBlocks, 128, 0, stream>>>(aggb, wlb, featb, wrb, b1, h1, N_NODES);

    // ---- SAGEConv 2 ----
    zero_kernel<<<zeroBlocks, 256, 0, stream>>>(agg, (size_t)N_NODES * CH);
    scatter_kernel<<<scatterBlocks, 256, 0, stream>>>(src, dst, h1, agg, nullptr, N_EDGES);
    cvt_bf16_kernel<<<cvtBlocks, 256, 0, stream>>>(h1,  nullptr, (unsigned*)featb, N_NODES);
    cvt_bf16_kernel<<<cvtBlocks, 256, 0, stream>>>(agg, deg,     (unsigned*)aggb,  N_NODES);
    cvt_bf16_kernel<<<wcvtBlocks, 256, 0, stream>>>(W2l, nullptr, (unsigned*)wlb, CH);
    cvt_bf16_kernel<<<wcvtBlocks, 256, 0, stream>>>(W2r, nullptr, (unsigned*)wrb, CH);
    sage_gemm_kernel<<<gemmBlocks, 128, 0, stream>>>(aggb, wlb, featb, wrb, b2, h2, N_NODES);

    // ---- global add pool ----
    zero_kernel<<<zeroBlocks, 256, 0, stream>>>(pooled, (size_t)NGRAPH * CH);
    pool_kernel<<<poolBlocks, 256, 0, stream>>>(h2, batch, pooled, N_NODES);

    // ---- MLP head ----
    cvt_bf16_kernel<<<(NGRAPH * CH / 2 + 255) / 256, 256, 0, stream>>>(pooled, nullptr,
                                                                       (unsigned*)pooledb, NGRAPH);
    cvt_bf16_kernel<<<wcvtBlocks, 256, 0, stream>>>(l1w, nullptr, (unsigned*)l1wb, CH);
    lin1_bn_relu_kernel<<<headBlocks, 128, 0, stream>>>(pooledb, l1wb, l1b, gam, bet,
                                                        rmean, rvar, hg, NGRAPH);
    lin2_kernel<<<(NGRAPH * 2 + 255) / 256, 256, 0, stream>>>(hg, l2w, l2b, (float*)d_out);
}
